// RepLKBlock3D_12240656793621
// MI455X (gfx1250) — matrix operand order repaired and checked by reading, not compiled, not run on hardware
//
#include <hip/hip_runtime.h>

// ---------------------------------------------------------------------------
// RepLK 3D depthwise block on gfx1250 (MI455X), Toeplitz-WMMA formulation.
//
//   out = relu(x + relu(bn7(dw7(x))) + relu(bn3(dw3(x))) + relu(bn1(x*w1)))
//
// 1D conv along W as banded-Toeplitz matmul per (kd,kh) tap plane:
//   A[m,k] = w7[c,kd,kh,k-m]   (0 <= k-m <= 6, else 0)     -> 16x32 bf16
//   B[k,n] = xtile[plane p, h0+n+kh-3, w0-3+k]             -> 32x16 bf16
//   acc7[dd][m,n] += A(kd=p-dd,kh) x B(p,kh)   (fp32 accum)
//
// Operand-reuse schedule (this revision): all 4 D-slice accumulators live at
// once; per kh the 7 A tiles are register-cached; per (p,kh) one B load feeds
// up to 4 WMMAs for the 7^3 conv plus the shared-B 3^3-conv WMMAs.
// LDS traffic: ~1.1 ds_load_b128 per WMMA (was ~3.7).
// ---------------------------------------------------------------------------

typedef __attribute__((ext_vector_type(16))) __bf16 v16bf;
typedef __attribute__((ext_vector_type(8)))  float  v8f;

#define C_   32
#define D_   64
#define H_   128
#define W_   128
#define DT   4      // D outputs per block (all 4 accumulated concurrently)
#define HT   16     // H outputs per block (== WMMA N)
// LDS input tile: planes d0-3..d0+6, rows h0-3..h0+18, w elements -3..140.
// Row pitch 304 B (76 dw == 12 mod 64 banks) -> conflict-free row-strided loads.
#define EW   144                 // valid bf16 elements per row
#define RPB  304                 // row pitch bytes (16B aligned, bank-friendly)
#define HH   22
#define PPB  (HH * RPB)          // 6688 B plane pitch
#define DD   10
#define IN_BYTES (DD * PPB)      // 66880
#define A7_OFF   IN_BYTES        // 49 Toeplitz tiles, 1 KB each (16x32 bf16)
#define A7_BYTES (49 * 1024)
#define A3_OFF   (A7_OFF + A7_BYTES)
#define A3_BYTES (9 * 1024)
#define LDS_BYTES (A3_OFF + A3_BYTES)   // 126272 B -> 2 WGs / WGP (320 KB LDS)

__device__ __forceinline__ unsigned short f2bf(float f) {
    union { float f; unsigned u; } v; v.f = f;
    unsigned r = v.u + 0x7FFFu + ((v.u >> 16) & 1u);   // round-to-nearest-even
    return (unsigned short)(r >> 16);
}

// 32 B operand fetch as two 16B-aligned ds_load_b128 (row pitch is not 32B-mult).
__device__ __forceinline__ v16bf ld_op(const char* p) {
    union { uint4 q[2]; v16bf v; } u;
    u.q[0] = *(const uint4*)(p);
    u.q[1] = *(const uint4*)(p + 16);
    return u.v;
}

__global__ __launch_bounds__(256)
void replk3d_wmma(const float* __restrict__ x,
                  const float* __restrict__ w7, const float* __restrict__ w3,
                  const float* __restrict__ w1,
                  const float* __restrict__ g7, const float* __restrict__ b7,
                  const float* __restrict__ m7, const float* __restrict__ v7,
                  const float* __restrict__ g3, const float* __restrict__ b3,
                  const float* __restrict__ m3, const float* __restrict__ v3,
                  const float* __restrict__ g1, const float* __restrict__ b1,
                  const float* __restrict__ m1, const float* __restrict__ v1,
                  float* __restrict__ out)
{
    extern __shared__ char smem[];
    const int tid  = threadIdx.x;
    const int blk  = blockIdx.x;
    const int hblk = blk & 7;             // H/HT = 8
    const int dblk = (blk >> 3) & 15;     // D/DT = 16
    const int c    = (blk >> 7) & 31;     // C = 32
    const int ni   = blk >> 12;           // N = 2
    const int h0   = hblk * HT;
    const int d0   = dblk * DT;

    // Folded BatchNorm (uniform per block -> scalar regs).
    const float inv7 = g7[c] * rsqrtf(v7[c] + 1e-5f);
    const float bb7  = b7[c] - m7[c] * inv7;
    const float inv3 = g3[c] * rsqrtf(v3[c] + 1e-5f);
    const float bb3  = b3[c] - m3[c] * inv3;
    const float inv1 = g1[c] * rsqrtf(v1[c] + 1e-5f);
    const float a1   = w1[c] * inv1;               // 1x1x1 dwconv folded into BN
    const float bb1  = b1[c] - m1[c] * inv1;

    // ---- Stage 1: haloed input tile -> LDS (fp32 -> bf16; zero fill = conv pad)
    const float* xc = x + ((size_t)ni * C_ + c) * (size_t)(D_ * H_ * W_);
    for (int idx = tid; idx < DD * HH * EW; idx += 256) {
        int e  = idx % EW;
        int r  = idx / EW;
        int hh = r % HH;
        int pp = r / HH;
        int d = d0 + pp - 3, h = h0 + hh - 3, w = e - 3;
        float val = 0.0f;
        if ((unsigned)d < (unsigned)D_ && (unsigned)h < (unsigned)H_ &&
            (unsigned)w < (unsigned)W_)
            val = xc[((size_t)d * H_ + h) * W_ + w];
        *(unsigned short*)(smem + pp * PPB + hh * RPB + e * 2) = f2bf(val);
    }

    // ---- Stage 2: Toeplitz A tiles in this kernel's per-lane A operand order
    // (contiguous k, as B uses): lane L (m = L&15) owns 32 contiguous bytes = 16 bf16 slots;
// slot s maps to K = s + (L<16?0:16).
    const float* w7c = w7 + c * 343;
    for (int idx = tid; idx < 49 * 32 * 16; idx += 256) {
        int s = idx & 15, L = (idx >> 4) & 31, t = idx >> 9;   // t = kd*7+kh
        int m = L & 15;
        int K = s + (L & 16);
        int j = K - m;                                          // tap index kw
        unsigned short val = (j >= 0 && j <= 6) ? f2bf(w7c[t * 7 + j]) : 0;
        *(unsigned short*)(smem + A7_OFF + t * 1024 + L * 32 + s * 2) = val;
    }
    const float* w3c = w3 + c * 27;
    for (int idx = tid; idx < 9 * 32 * 16; idx += 256) {
        int s = idx & 15, L = (idx >> 4) & 31, t = idx >> 9;   // t = kd3*3+kh3
        int m = L & 15;
        int K = s + (L & 16);
        int j = K - m - 2;                                      // +2: shares 7-tap B
        unsigned short val = (j >= 0 && j <= 2) ? f2bf(w3c[t * 3 + j]) : 0;
        *(unsigned short*)(smem + A3_OFF + t * 1024 + L * 32 + s * 2) = val;
    }
    __syncthreads();

    // ---- Stage 3: WMMA sweep. Wave wv owns W column [16*wv, 16*wv+16).
    const int lane = tid & 31;
    const int wv   = tid >> 5;            // 0..7
    const int n16  = lane & 15;           // B/D column = h row
    const int hiL  = lane >> 4;           // lane half -> K/M split
    const int w0   = wv * 16;

    const char* aBase  = smem + A7_OFF + lane * 32;
    const char* a3Base = smem + A3_OFF + lane * 32;
    // B 32x16 bf16: lanes 0-15 hold K=0..15 (w0..w0+15), lanes 16-31 K=16..31.
    const char* bBase  = smem + n16 * RPB + w0 * 2 + hiL * 32;

    // Warm L0 for the fp32 residual lines used by the epilogue.
    const size_t xrowBase =
        ((((size_t)ni * C_ + c) * D_ + d0) * H_ + (h0 + n16)) * (size_t)W_ +
        (w0 + hiL * 8);
    #pragma unroll
    for (int dd = 0; dd < DT; ++dd)
        __builtin_prefetch(x + xrowBase + (size_t)dd * H_ * W_, 0, 3);

    v8f acc7[DT] = {};
    v8f acc3[DT] = {};

    #pragma unroll
    for (int kh = 0; kh < 7; ++kh) {
        // Register-cache the 7 A tiles of this kh column (56 VGPRs).
        v16bf A[7];
        #pragma unroll
        for (int kd = 0; kd < 7; ++kd)
            A[kd] = ld_op(aBase + (kd * 7 + kh) * 1024);
        const bool centerH = (kh >= 2 && kh <= 4);
        v16bf A3t[3];
        if (centerH) {
            #pragma unroll
            for (int k3 = 0; k3 < 3; ++k3)
                A3t[k3] = ld_op(a3Base + (k3 * 3 + (kh - 2)) * 1024);
        }
        // One B load per input plane feeds up to 4 (dd,kd=p-dd) WMMAs,
        // plus the 3^3-conv WMMAs that share the same B operand.
        #pragma unroll
        for (int p = 0; p < DD; ++p) {
            v16bf Bv = ld_op(bBase + p * PPB + kh * RPB);
            #pragma unroll
            for (int dd = 0; dd < DT; ++dd) {
                const int kd = p - dd;
                if (kd >= 0 && kd <= 6) {
                    acc7[dd] = __builtin_amdgcn_wmma_f32_16x16x32_bf16(
                        false, A[kd], false, Bv, (short)0, acc7[dd], false, false);
                    if (centerH && kd >= 2 && kd <= 4)
                        acc3[dd] = __builtin_amdgcn_wmma_f32_16x16x32_bf16(
                            false, A3t[kd - 2], false, Bv, (short)0, acc3[dd],
                            false, false);
                }
            }
        }
    }

    // ---- Epilogue per D slice: D layout (ISA): VGPR r -> M = r + 8*hiL (w), N = n16 (h).
    #pragma unroll
    for (int dd = 0; dd < DT; ++dd) {
        const size_t base = xrowBase + (size_t)dd * H_ * W_;
        const float4 x0 = *(const float4*)(x + base);
        const float4 x1 = *(const float4*)(x + base + 4);
        float o[8];
        #pragma unroll
        for (int r = 0; r < 8; ++r) {
            float xr = (r < 4) ? ((const float*)&x0)[r] : ((const float*)&x1)[r - 4];
            float y7 = fmaxf(inv7 * acc7[dd][r] + bb7, 0.0f);
            float y3 = fmaxf(inv3 * acc3[dd][r] + bb3, 0.0f);
            float y1 = fmaxf(a1 * xr + bb1, 0.0f);
            o[r] = fmaxf(xr + y7 + y3 + y1, 0.0f);
        }
        *(float4*)(out + base)     = make_float4(o[0], o[1], o[2], o[3]);
        *(float4*)(out + base + 4) = make_float4(o[4], o[5], o[6], o[7]);
    }
}

extern "C" void kernel_launch(void* const* d_in, const int* in_sizes, int n_in,
                              void* d_out, int out_size, void* d_ws, size_t ws_size,
                              hipStream_t stream) {
    (void)in_sizes; (void)n_in; (void)out_size; (void)d_ws; (void)ws_size;
    const float* x  = (const float*)d_in[0];
    const float* w7 = (const float*)d_in[1];
    const float* w3 = (const float*)d_in[2];
    const float* w1 = (const float*)d_in[3];
    const float* g7 = (const float*)d_in[4];
    const float* b7 = (const float*)d_in[5];
    const float* m7 = (const float*)d_in[6];
    const float* v7 = (const float*)d_in[7];
    const float* g3 = (const float*)d_in[8];
    const float* b3 = (const float*)d_in[9];
    const float* m3 = (const float*)d_in[10];
    const float* v3 = (const float*)d_in[11];
    const float* g1 = (const float*)d_in[12];
    const float* b1 = (const float*)d_in[13];
    const float* m1 = (const float*)d_in[14];
    const float* v1 = (const float*)d_in[15];
    float* out = (float*)d_out;

    const int nblocks = 2 * C_ * (D_ / DT) * (H_ / HT);   // 8192
    replk3d_wmma<<<dim3(nblocks), dim3(256), LDS_BYTES, stream>>>(
        x, w7, w3, w1, g7, b7, m7, v7, g3, b3, m3, v3, g1, b1, m1, v1, out);
}
